// RadixAttention_83090437308961
// MI455X (gfx1250) — matrix operand order repaired and checked by reading, not compiled, not run on hardware
//
#include <hip/hip_runtime.h>
#include <cstdint>

typedef __attribute__((ext_vector_type(16))) __bf16 v16bf;
typedef __attribute__((ext_vector_type(8)))  float  v8f;

#define SQ   2048
#define NH   32
#define NKV  8
#define DH   128
#define BM   64
#define BN   32
#define KSTR 40    // K LDS row stride (halves): 80B rows (64B data + pad), 16B aligned
#define VSTR 136   // V LDS row stride (halves): 272B rows (256B data + pad), 16B aligned
#define PSTR 34    // P staging row stride (halves)

#define KT_ELEMS   (2ull * NKV * 64 * 4096)   // bf16 K, transposed 32-key tiles [d][n]
// ws layout: [0, KT_ELEMS) = kT ; [KT_ELEMS, 2*KT_ELEMS) = v (bf16, original layout)

__device__ __forceinline__ v16bf lds_ld16bf(const __bf16* p) {
  union { uint4 u[2]; v16bf v; } t;
  t.u[0] = *reinterpret_cast<const uint4*>(p);
  t.u[1] = *reinterpret_cast<const uint4*>(p + 8);
  return t.v;
}

__device__ __forceinline__ void async_b128(uint32_t lds_off, const void* g) {
  asm volatile("global_load_async_to_lds_b128 %0, %1, off"
               :: "v"(lds_off), "v"((uint64_t)(uintptr_t)g) : "memory");
}

// ---- pre-pass 1: V f32 -> bf16, same [b][s][kvh][d] layout (rows stay contiguous) ----
__global__ __launch_bounds__(256)
void prep_v_bf16(const float* __restrict__ in, __bf16* __restrict__ out) {
  const int i = blockIdx.x * 256 + threadIdx.x;          // pair index
  const float2 f = reinterpret_cast<const float2*>(in)[i];
  union { __bf16 h[2]; uint32_t u; } t;
  t.h[0] = (__bf16)f.x; t.h[1] = (__bf16)f.y;
  reinterpret_cast<uint32_t*>(out)[i] = t.u;
}

// ---- pre-pass 2: K f32 -> bf16, transposed per 32-key tile: kT[(b,kvh,t)][d][n] ----
__global__ __launch_bounds__(256)
void prep_k_transpose(const float* __restrict__ k, __bf16* __restrict__ kT) {
  __shared__ __bf16 tl[32 * 132];
  const int t   = blockIdx.x & 63;
  const int kvh = (blockIdx.x >> 6) & 7;
  const int b   = blockIdx.x >> 9;
  const int tid = threadIdx.x;
  const float* kin = k + (((size_t)b * SQ + (size_t)t * 32) * NKV + kvh) * DH;
  for (int idx = tid; idx < 32 * 128; idx += 256) {
    const int n = idx >> 7, d = idx & 127;
    tl[n * 132 + d] = (__bf16)kin[(size_t)n * (NKV * DH) + d];   // coalesced over d
  }
  __syncthreads();
  __bf16* outp = kT + (size_t)blockIdx.x * 4096;
  for (int idx = tid; idx < 4096; idx += 256) {
    const int d = idx >> 5, n = idx & 31;
    outp[idx] = tl[n * 132 + d];                                 // coalesced store
  }
}

// ---- main: causal GQA flash attention, bf16 WMMA, async double-buffered K/V ----
__global__ __launch_bounds__(128, 1)
void fa_fwd_kernel(const float* __restrict__ q, const __bf16* __restrict__ kT,
                   const __bf16* __restrict__ vB, float* __restrict__ out) {
  __shared__ alignas(16) __bf16 Kbuf[2][DH * KSTR];
  __shared__ alignas(16) __bf16 Vbuf[2][BN * VSTR];
  __shared__ alignas(16) __bf16 Plds[4 * 16 * PSTR];

  const int tid  = threadIdx.x;
  const int lane = tid & 31;
  const int wv   = tid >> 5;
  const int qx   = blockIdx.x;
  const int h    = blockIdx.y;
  const int b    = blockIdx.z;
  const int kvh  = h >> 2;                 // G = 4

  const int qbase = qx * BM;
  const int m  = lane & 15;
  const int hi = lane >> 4;
  const float scale = 0.08838834764831845f;  // 1/sqrt(128)

  // Q -> A fragments (pre-scaled bf16), once per block
  const int qrow_a = qbase + wv * 16 + m;
  const float* qp = q + (((size_t)b * SQ + qrow_a) * NH + h) * DH;
  v16bf qa[4];
  #pragma unroll
  for (int c = 0; c < 4; ++c) {
    #pragma unroll
    for (int vg = 0; vg < 8; ++vg) {
      const int kk = (vg < 4 ? 2 * vg : 16 + 2 * (vg - 4)) + hi * 8;
      const float2 f2 = *reinterpret_cast<const float2*>(qp + c * 32 + kk);
      qa[c][2 * vg]     = (__bf16)(f2.x * scale);
      qa[c][2 * vg + 1] = (__bf16)(f2.y * scale);
    }
  }

  v8f oacc[8] = {};
  float mrow[8], lsum[8];
  #pragma unroll
  for (int r = 0; r < 8; ++r) { mrow[r] = -INFINITY; lsum[r] = 0.0f; }

  const int row_c = qbase + wv * 16 + 8 * hi;
  const int nkb = 2 * qx + 2;
  const __bf16* kTh = kT + (size_t)(b * NKV + kvh) * 64 * 4096;
  const __bf16* vh  = vB + ((size_t)b * SQ * NKV + kvh) * (size_t)DH;
  __bf16* pw = Plds + wv * 16 * PSTR;

  // issue one K+V tile's async copies (8 x b128 per thread)
  auto issue_tile = [&](int kbi, int bufi) {
    const __bf16* kt = kTh + (size_t)kbi * 4096;
    const __bf16* vt = vh + (size_t)(kbi * BN) * (NKV * DH);
    const uint32_t kl = (uint32_t)(uintptr_t)(&Kbuf[bufi][0]);
    const uint32_t vl = (uint32_t)(uintptr_t)(&Vbuf[bufi][0]);
    #pragma unroll
    for (int it = 0; it < 4; ++it) {
      const int idx = tid + it * 128;          // 0..511
      const int d = idx >> 2, c = idx & 3;
      async_b128(kl + d * 80 + c * 16, (const char*)kt + (size_t)idx * 16);
    }
    #pragma unroll
    for (int it = 0; it < 4; ++it) {
      const int idx = tid + it * 128;          // 0..511
      const int n = idx >> 4, c = idx & 15;
      async_b128(vl + n * 272 + c * 16, (const char*)vt + (size_t)n * (NKV * DH * 2) + c * 16);
    }
  };

  issue_tile(0, 0);

  for (int kb = 0; kb < nkb; ++kb) {
    const int key0 = kb * BN;
    const int buf = kb & 1;

    // prefetch next tile while current finishes; async loads complete in order,
    // so <=8 outstanding means the current tile's 8 chunks have landed.
    if (kb + 1 < nkb) {
      issue_tile(kb + 1, buf ^ 1);
      asm volatile("s_wait_asynccnt 8" ::: "memory");
    } else {
      asm volatile("s_wait_asynccnt 0" ::: "memory");
    }
    __syncthreads();

    const __bf16* Kb = Kbuf[buf];
    const __bf16* Vb = Vbuf[buf];

    // ---- S = Q * K^T ----
    v8f s0 = {}, s1 = {};
    #pragma unroll
    for (int c = 0; c < 4; ++c) {
      v16bf kf0, kf1;
      #pragma unroll
      for (int s = 0; s < 16; ++s) {
        const int kk = (s < 8 ? s : 8 + s) + hi * 8;
        kf0[s] = Kb[(c * 32 + kk) * KSTR + m];
        kf1[s] = Kb[(c * 32 + kk) * KSTR + 16 + m];
      }
      s0 = __builtin_amdgcn_wmma_f32_16x16x32_bf16(false, qa[c], false, kf0, (short)0, s0, false, false);
      s1 = __builtin_amdgcn_wmma_f32_16x16x32_bf16(false, qa[c], false, kf1, (short)0, s1, false, false);
    }

    // ---- online softmax (C layout), stage P to LDS ----
    const bool needMask = (key0 + BN - 1) > (qbase + wv * 16);
    #pragma unroll
    for (int r = 0; r < 8; ++r) {
      float x0 = s0[r], x1 = s1[r];
      const int row = row_c + r;
      if (needMask) {
        const int n0 = key0 + m;
        if (n0 > row)      x0 = -1e30f;
        if (n0 + 16 > row) x1 = -1e30f;
      }
      float mx = fmaxf(x0, x1);
      #pragma unroll
      for (int off = 1; off < 16; off <<= 1) mx = fmaxf(mx, __shfl_xor(mx, off, 32));
      const float mn = fmaxf(mrow[r], mx);
      const float al = __expf(mrow[r] - mn);
      const float p0 = __expf(x0 - mn);
      const float p1 = __expf(x1 - mn);
      float ps = p0 + p1;
      #pragma unroll
      for (int off = 1; off < 16; off <<= 1) ps += __shfl_xor(ps, off, 32);
      lsum[r] = lsum[r] * al + ps;
      mrow[r] = mn;
      pw[(8 * hi + r) * PSTR + m]      = (__bf16)p0;
      pw[(8 * hi + r) * PSTR + 16 + m] = (__bf16)p1;
      #pragma unroll
      for (int t = 0; t < 8; ++t) oacc[t][r] *= al;
    }

    asm volatile("s_wait_dscnt 0" ::: "memory");   // wave-local LDS RAW

    // ---- P: C layout -> A fragment ----
    v16bf pa;
    #pragma unroll
    for (int vg = 0; vg < 8; ++vg) {
      const int kk = (vg < 4 ? 2 * vg : 16 + 2 * (vg - 4)) + hi * 8;
      union { uint32_t u; __bf16 hh[2]; } tu;
      tu.u = *reinterpret_cast<const uint32_t*>(&pw[m * PSTR + kk]);
      pa[2 * vg]     = tu.hh[0];
      pa[2 * vg + 1] = tu.hh[1];
    }

    // ---- O += P * V ----
    #pragma unroll
    for (int t = 0; t < 8; ++t) {
      v16bf vf;
      #pragma unroll
      for (int s = 0; s < 16; ++s) {
        const int kk = (s < 8 ? s : 8 + s) + hi * 8;
        vf[s] = Vb[kk * VSTR + t * 16 + m];
      }
      oacc[t] = __builtin_amdgcn_wmma_f32_16x16x32_bf16(false, pa, false, vf, (short)0, oacc[t], false, false);
    }

    __syncthreads();   // all waves done with buf before it is refilled
  }

  // ---- epilogue ----
  #pragma unroll
  for (int r = 0; r < 8; ++r) {
    const float inv = 1.0f / lsum[r];
    const int row = row_c + r;
    float* op = out + (((size_t)b * SQ + row) * NH + h) * DH + m;
    #pragma unroll
    for (int t = 0; t < 8; ++t) op[t * 16] = oacc[t][r] * inv;
  }
}

extern "C" void kernel_launch(void* const* d_in, const int* in_sizes, int n_in,
                              void* d_out, int out_size, void* d_ws, size_t ws_size,
                              hipStream_t stream) {
  (void)in_sizes; (void)n_in; (void)out_size; (void)ws_size;
  const float* q = (const float*)d_in[0];
  const float* k = (const float*)d_in[1];
  const float* v = (const float*)d_in[2];
  float* out = (float*)d_out;

  __bf16* wsK = (__bf16*)d_ws;                 // transposed bf16 K tiles
  __bf16* wsV = wsK + KT_ELEMS;                // bf16 V, original layout

  prep_k_transpose<<<dim3(2 * NKV * 64), dim3(256), 0, stream>>>(k, wsK);
  prep_v_bf16<<<dim3((2 * SQ * NKV * DH / 2) / 256), dim3(256), 0, stream>>>(v, wsV);

  fa_fwd_kernel<<<dim3(SQ / BM, NH, 2), dim3(128), 0, stream>>>(q, wsK, wsV, out);
}
